// InstrumentEnhancedGenerator_72241349918969
// MI455X (gfx1250) — matrix operand order repaired and checked by reading, not compiled, not run on hardware
//
#include <hip/hip_runtime.h>
#include <hip/hip_bf16.h>

typedef __attribute__((ext_vector_type(16))) _Float16 v16h;
typedef __attribute__((ext_vector_type(8)))  _Float16 v8h;
typedef __attribute__((ext_vector_type(8)))  float    v8f;

// ---- problem constants ----
#define SRATE   22050.0f
#define LTOT    2097153      // output length == NFFT/2+1
#define NFFT    4194304      // 4^11
#define NQ      (NFFT/4)
#define LC1     1048577      // conv1 out length
#define LC2     524289       // conv2 out length
#define LC3     262145       // conv3 out length
#define K2TOT   160          // 32 ic * 5
#define K3TOT   320          // 64 ic * 5

// ---- workspace byte offsets ----
#define OFF_A    0ULL                 // float2[NFFT]  32MiB
#define OFF_B    33554432ULL          // float2[NFFT]  32MiB
#define OFF_CM   67108864ULL          // float[LTOT]   (dead before conv1)
#define OFF_BASE 79691776ULL          // float[LTOT]
#define OFF_H1   0ULL                 // _Float16[32*LC1]  (reuses FFT bufs)
#define OFF_H2   93323264ULL          // _Float16[64*LC2]
#define OFF_SC   167772160ULL         // small scratch: [0]=acc, [64..191]=feat
#define OFF_W2   (OFF_SC + 4096ULL)   // _Float16[64*160]
#define OFF_W3   (OFF_W2 + 20480ULL)  // _Float16[128*320]

__device__ __forceinline__ float2 cmulf(float2 v, float c, float s) {
    return make_float2(v.x * c - v.y * s, v.x * s + v.y * c);
}

// ---------------- init: pad signal to complex, cm=1, zero scratch ----------------
__global__ void k_init(const float* __restrict__ z, float2* __restrict__ A,
                       float* __restrict__ cm, float* __restrict__ scr) {
    int i = blockIdx.x * blockDim.x + threadIdx.x;
    if (i < NFFT) A[i] = make_float2(i < LTOT ? z[i] : 0.0f, 0.0f);
    if (i < LTOT) cm[i] = 1.0f;
    if (i < 256)  scr[i] = 0.0f;
}

// ---------------- prog_mean: tiny MLP reduced over all t ----------------
__global__ __launch_bounds__(256) void k_progmean(
    const float* __restrict__ w1, const float* __restrict__ b1,
    const float* __restrict__ w2, const float* __restrict__ b2,
    const float* __restrict__ w3, const float* __restrict__ b3,
    const float* __restrict__ hp, float* __restrict__ acc) {
    __shared__ float sw1[32], sb1[32], sw2[512], sb2[16], sw3[16], shp[8], sb3;
    __shared__ float sred[256];
    int tid = threadIdx.x;
    if (tid < 32)  { sw1[tid] = w1[tid]; sb1[tid] = b1[tid]; }
    if (tid < 16)  { sb2[tid] = b2[tid]; sw3[tid] = w3[tid * 8 + 1]; }
    if (tid < 8)     shp[tid] = 0.25f * (hp[4*tid] + hp[4*tid+1] + hp[4*tid+2] + hp[4*tid+3]);
    if (tid == 0)    sb3 = b3[1];
    for (int i = tid; i < 512; i += 256) sw2[i] = w2[i];
    __syncthreads();
    const float invL = 1.0f / (float)LTOT;
    float partial = 0.0f;
    int stride = gridDim.x * blockDim.x;
    for (int t = blockIdx.x * blockDim.x + tid; t < LTOT; t += stride) {
        float tn = (float)t * invL;
        float h1v[32];
        #pragma unroll
        for (int j = 0; j < 32; ++j) {
            float v = tn * sw1[j] + sb1[j];
            h1v[j] = v > 0.0f ? v : 0.0f;
        }
        float mp1 = sb3;
        #pragma unroll
        for (int k = 0; k < 16; ++k) {
            float a = sb2[k];
            #pragma unroll
            for (int j = 0; j < 32; ++j) a += h1v[j] * sw2[j * 16 + k];
            a = a > 0.0f ? a : 0.0f;
            mp1 += a * sw3[k];
        }
        float sn = __sinf(6.2831853f * tn);
        int c = (int)(tn * 8.0f);
        c = c < 0 ? 0 : (c > 7 ? 7 : c);
        partial += (1.0f + mp1 * sn) * shp[c];
    }
    sred[tid] = partial;
    __syncthreads();
    for (int s = 128; s > 0; s >>= 1) { if (tid < s) sred[tid] += sred[tid + s]; __syncthreads(); }
    if (tid == 0) atomicAdd(acc, sred[0]);
}

// ---------------- band multipliers into cm ----------------
__global__ void k_bandmult(float* __restrict__ cm, const float* __restrict__ bw) {
    int i = blockIdx.x * blockDim.x + threadIdx.x;
    if (i >= LTOT) return;
    const float lo[6] = {1.f, 4.f, 8.f, 13.f, 30.f, 100.f};
    const float hi[6] = {4.f, 8.f, 13.f, 30.f, 100.f, 200.f};
    float f = (float)i * (SRATE / (float)NFFT);
    float m = 1.0f;
    #pragma unroll
    for (int b = 0; b < 6; ++b) {
        float c  = 0.5f * (lo[b] + hi[b]);
        float sg = 0.25f * (hi[b] - lo[b]);           // (w/2) with w=(hi-lo)/2
        if (f >= lo[b] && f <= hi[b]) {
            float d = (f - c) / sg;
            float mask = expf(-0.5f * d * d);
            double ang = 6.283185307179586 * (double)c * (double)i / 22050.0;
            float tmod = (float)sin(ang);
            m *= 1.0f + mask * bw[b] * (1.0f + 0.2f * tmod);
        }
    }
    cm[i] *= m;
}

// ---------------- specific-frequency windows (single wave; overlaps are same-lane) --------
__global__ void k_specwin(float* __restrict__ cm, const float* __restrict__ fw,
                          const float* __restrict__ acc) {
    const float specs[8] = {7.83f, 528.f, 396.f, 2.5f, 14.1f, 432.f, 6.f, 30.f};
    float prog_mean = acc[0] * (1.0f / (float)LTOT);
    int lane = threadIdx.x;
    for (int k = 0; k < 8; ++k) {
        for (int m = 1; m <= 5; ++m) {
            float hf = specs[k] * (float)m;
            if (hf >= 0.5f * SRATE) continue;
            int hidx = (int)roundf(hf * ((float)NFFT / SRATE));
            int s = hidx - 15; if (s < 0) s = 0;
            int e = hidx + 15; if (e > LTOT - 1) e = LTOT - 1;
            int idx = s + lane;
            if (idx <= e) {
                float d = (float)(idx - hidx);
                float win = expf(-0.5f * d * d * (1.0f / 25.0f));   // sigma = WIN/3 = 5
                float enh = fw[k] * win * powf((float)m, -1.2f) * (1.0f + prog_mean);
                cm[idx] *= (1.0f + enh);
            }
        }
    }
}

// ---------------- Stockham radix-4 FFT pass (dir=+1 fwd e^{-i}, dir=-1 inverse) ----------
__global__ __launch_bounds__(256) void k_fft4(const float2* __restrict__ x,
                                              float2* __restrict__ y,
                                              int Ns, int lg, float dir) {
    int j = blockIdx.x * blockDim.x + threadIdx.x;
    if (j >= NQ) return;
    int p = j & (Ns - 1);
    float2 v0 = x[j];
    float2 v1 = x[j + NQ];
    float2 v2 = x[j + 2 * NQ];
    float2 v3 = x[j + 3 * NQ];
    float ang = dir * (-6.2831853071795864f) * (float)p / (float)(4 * Ns);
    float s1, c1; __sincosf(ang, &s1, &c1);
    float c2 = c1 * c1 - s1 * s1, s2 = 2.0f * c1 * s1;
    float c3 = c2 * c1 - s2 * s1, s3 = c2 * s1 + s2 * c1;
    v1 = cmulf(v1, c1, s1);
    v2 = cmulf(v2, c2, s2);
    v3 = cmulf(v3, c3, s3);
    float2 t0 = make_float2(v0.x + v2.x, v0.y + v2.y);
    float2 t1 = make_float2(v0.x - v2.x, v0.y - v2.y);
    float2 t2 = make_float2(v1.x + v3.x, v1.y + v3.y);
    float2 t3 = make_float2(v1.x - v3.x, v1.y - v3.y);
    float2 it3 = make_float2(-t3.y, t3.x);
    int idxD = ((j >> lg) << (lg + 2)) | p;
    y[idxD]          = make_float2(t0.x + t2.x, t0.y + t2.y);
    y[idxD + Ns]     = make_float2(t1.x - dir * it3.x, t1.y - dir * it3.y);
    y[idxD + 2 * Ns] = make_float2(t0.x - t2.x, t0.y - t2.y);
    y[idxD + 3 * Ns] = make_float2(t1.x + dir * it3.x, t1.y + dir * it3.y);
}

// ------- apply combined multiplier + 3-tap magnitude smoothing; emit Hermitian spectrum ----
__global__ void k_apply(const float2* __restrict__ S, float2* __restrict__ D,
                        const float* __restrict__ cm) {
    int i = blockIdx.x * blockDim.x + threadIdx.x;
    if (i >= LTOT) return;
    float2 Xi = S[i];
    float cmi = cm[i];
    float2 Yi = make_float2(Xi.x * cmi, Xi.y * cmi);
    float2 Z;
    if (i == 0 || i == LTOT - 1) {
        Z = Yi;
    } else {
        float2 Xm = S[i - 1]; float cmm = cm[i - 1];
        float2 Xp = S[i + 1]; float cmp_ = cm[i + 1];
        float mi = sqrtf(Yi.x * Yi.x + Yi.y * Yi.y);
        float mm = sqrtf(Xm.x * Xm.x + Xm.y * Xm.y) * cmm;
        float mp = sqrtf(Xp.x * Xp.x + Xp.y * Xp.y) * cmp_;
        float ms = 0.7f * mi + 0.15f * mm + 0.15f * mp;
        if (mi > 1e-30f) {
            float r = ms / mi;
            Z = make_float2(Yi.x * r, Yi.y * r);
        } else {
            Z = make_float2(ms, 0.0f);
        }
    }
    D[i] = Z;
    if (i >= 1 && i <= LTOT - 2) D[NFFT - i] = make_float2(Z.x, -Z.y);
}

// ---------------- extract real part / N ----------------
__global__ void k_extract(const float2* __restrict__ S, float* __restrict__ base) {
    int i = blockIdx.x * blockDim.x + threadIdx.x;
    if (i >= LTOT) return;
    base[i] = S[i].x * (1.0f / (float)NFFT);
}

// ---------------- weight conversion to f16 ----------------
__global__ void k_wconv(const float* __restrict__ c2w, const float* __restrict__ c3w,
                        _Float16* __restrict__ W2, _Float16* __restrict__ W3) {
    int i = blockIdx.x * blockDim.x + threadIdx.x;
    if (i < 64 * K2TOT)  W2[i] = (_Float16)c2w[i];
    if (i < 128 * K3TOT) W3[i] = (_Float16)c3w[i];
}

// ---------------- conv1: 1->32 ch, scalar (K=5 too small for WMMA) ----------------
__global__ void k_conv1(const float* __restrict__ base, const float* __restrict__ w,
                        const float* __restrict__ b, _Float16* __restrict__ h1) {
    int t = blockIdx.x * blockDim.x + threadIdx.x;
    int oc = blockIdx.y;
    if (t >= LC1) return;
    float acc = b[oc];
    #pragma unroll
    for (int k = 0; k < 5; ++k) {
        int p = 2 * t + k - 2;
        if (p >= 0 && p < LTOT) acc += w[oc * 5 + k] * base[p];
    }
    acc = acc > 0.0f ? acc : 0.2f * acc;
    h1[oc * LC1 + t] = (_Float16)acc;
}

// ---------------- conv2: 32->64 ch via WMMA, M=64 (4 wave tiles), K=160 ----------------
// Stage 1: raw halo tile (coalesced). Stage 2: im2col into per-lane fragment order so
// each B fragment is one 32B-aligned contiguous v16h LDS load (2x ds_load_b128).
__global__ __launch_bounds__(128) void k_conv2(const _Float16* __restrict__ h1,
                                               const _Float16* __restrict__ W,
                                               const float* __restrict__ bias,
                                               _Float16* __restrict__ h2) {
    __shared__ _Float16 sIn[32][38];
    __shared__ alignas(32) _Float16 sB[16][K2TOT];
    int t0 = blockIdx.x * 16;
    int tid = threadIdx.x;
    for (int i = tid; i < 32 * 36; i += 128) {
        int ic = i / 36, pp = i - ic * 36;
        int gp = 2 * t0 - 2 + pp;
        _Float16 v = (_Float16)0.0f;
        if (gp >= 0 && gp < LC1) v = h1[ic * LC1 + gp];
        sIn[ic][pp] = v;
    }
    __syncthreads();
    for (int i = tid; i < 16 * K2TOT; i += 128) {
        int n = i / K2TOT, kk = i - n * K2TOT;
        int ic = kk / 5, k5 = kk - 5 * ic;
        sB[n][kk] = sIn[ic][2 * n + k5];
    }
    __syncthreads();
    int lane  = tid & 31;
    int oc0   = (tid >> 5) * 16;
    int mrow  = lane & 15;
    int ahalf = (lane >> 4) * 8;
    int bn    = lane & 15;
    int bks   = (lane >> 4) * 16;
    const _Float16* wrow = &W[(oc0 + mrow) * K2TOT];
    __builtin_prefetch(wrow, 0, 3);
    v8f c = {};
    #pragma unroll
    for (int kb = 0; kb < K2TOT; kb += 32) {
        v8h alo = *(const v8h*)(wrow + kb + 2 * ahalf);
        v8h ahi = *(const v8h*)(wrow + kb + 2 * ahalf + 8);
        v16h a = __builtin_shufflevector(alo, ahi, 0, 1, 2, 3, 4, 5, 6, 7,
                                                   8, 9, 10, 11, 12, 13, 14, 15);
        v16h b = *(const v16h*)(&sB[bn][kb + bks]);
        c = __builtin_amdgcn_wmma_f32_16x16x32_f16(false, a, false, b, (short)0, c, false, false);
    }
    int t = t0 + bn;
    if (t < LC2) {
        #pragma unroll
        for (int r = 0; r < 8; ++r) {
            int oc = oc0 + ahalf + r;
            float v = c[r] + bias[oc];
            v = v > 0.0f ? v : 0.2f * v;
            h2[oc * LC2 + t] = (_Float16)v;
        }
    }
}

// ---------------- conv3: 64->128 ch via WMMA (8 wave tiles), fused channel-mean ----------
__global__ __launch_bounds__(256) void k_conv3(const _Float16* __restrict__ h2,
                                               const _Float16* __restrict__ W,
                                               const float* __restrict__ bias,
                                               float* __restrict__ gFeat) {
    __shared__ _Float16 sIn[64][38];
    __shared__ alignas(32) _Float16 sB[16][K3TOT];
    __shared__ float sAcc[128];
    int t0 = blockIdx.x * 16;
    int tid = threadIdx.x;
    if (tid < 128) sAcc[tid] = 0.0f;
    for (int i = tid; i < 64 * 36; i += 256) {
        int ic = i / 36, pp = i - ic * 36;
        int gp = 2 * t0 - 2 + pp;
        _Float16 v = (_Float16)0.0f;
        if (gp >= 0 && gp < LC2) v = h2[ic * LC2 + gp];
        sIn[ic][pp] = v;
    }
    __syncthreads();
    for (int i = tid; i < 16 * K3TOT; i += 256) {
        int n = i / K3TOT, kk = i - n * K3TOT;
        int ic = kk / 5, k5 = kk - 5 * ic;
        sB[n][kk] = sIn[ic][2 * n + k5];
    }
    __syncthreads();
    int lane  = tid & 31;
    int oc0   = (tid >> 5) * 16;
    int mrow  = lane & 15;
    int ahalf = (lane >> 4) * 8;
    int bn    = lane & 15;
    int bks   = (lane >> 4) * 16;
    const _Float16* wrow = &W[(oc0 + mrow) * K3TOT];
    __builtin_prefetch(wrow, 0, 3);
    v8f c = {};
    #pragma unroll
    for (int kb = 0; kb < K3TOT; kb += 32) {
        v8h alo = *(const v8h*)(wrow + kb + 2 * ahalf);
        v8h ahi = *(const v8h*)(wrow + kb + 2 * ahalf + 8);
        v16h a = __builtin_shufflevector(alo, ahi, 0, 1, 2, 3, 4, 5, 6, 7,
                                                   8, 9, 10, 11, 12, 13, 14, 15);
        v16h b = *(const v16h*)(&sB[bn][kb + bks]);
        c = __builtin_amdgcn_wmma_f32_16x16x32_f16(false, a, false, b, (short)0, c, false, false);
    }
    int t = t0 + bn;
    if (t < LC3) {
        #pragma unroll
        for (int r = 0; r < 8; ++r) {
            int oc = oc0 + ahalf + r;
            float v = c[r] + bias[oc];
            v = v > 0.0f ? v : 0.2f * v;
            atomicAdd(&sAcc[oc], v);
        }
    }
    __syncthreads();
    if (tid < 128) atomicAdd(&gFeat[tid], sAcc[tid]);
}

// ---------------- final MLP: feat/L3 (+64 zeros) -> 256 leaky -> scalar ----------------
__global__ __launch_bounds__(256) void k_final(const float* __restrict__ gFeat,
                                               const float* __restrict__ w1,
                                               const float* __restrict__ b1,
                                               const float* __restrict__ w2,
                                               const float* __restrict__ b2,
                                               float* __restrict__ out) {
    __shared__ float sF[128];
    __shared__ float sR[256];
    int tid = threadIdx.x;
    if (tid < 128) sF[tid] = gFeat[tid] * (1.0f / (float)LC3);
    __syncthreads();
    float h = b1[tid];
    for (int i = 0; i < 128; ++i) h += sF[i] * w1[i * 256 + tid];   // emb rows are zero
    h = h > 0.0f ? h : 0.2f * h;
    sR[tid] = h * w2[tid];
    __syncthreads();
    for (int s = 128; s > 0; s >>= 1) { if (tid < s) sR[tid] += sR[tid + s]; __syncthreads(); }
    if (tid == 0) out[0] = sR[0] + b2[0];
}

extern "C" void kernel_launch(void* const* d_in, const int* in_sizes, int n_in,
                              void* d_out, int out_size, void* d_ws, size_t ws_size,
                              hipStream_t stream) {
    const float* z    = (const float*)d_in[0];
    const float* bw   = (const float*)d_in[2];
    const float* fw   = (const float*)d_in[3];
    const float* harm = (const float*)d_in[4];
    const float* tw1  = (const float*)d_in[5];
    const float* tb1  = (const float*)d_in[6];
    const float* tw2  = (const float*)d_in[7];
    const float* tb2  = (const float*)d_in[8];
    const float* tw3  = (const float*)d_in[9];
    const float* tb3  = (const float*)d_in[10];
    const float* c1w  = (const float*)d_in[11];
    const float* c1b  = (const float*)d_in[12];
    const float* c2w  = (const float*)d_in[13];
    const float* c2b  = (const float*)d_in[14];
    const float* c3w  = (const float*)d_in[15];
    const float* c3b  = (const float*)d_in[16];
    const float* mw1  = (const float*)d_in[17];
    const float* mb1  = (const float*)d_in[18];
    const float* mw2  = (const float*)d_in[19];
    const float* mb2  = (const float*)d_in[20];

    char* ws = (char*)d_ws;
    float2*    A    = (float2*)(ws + OFF_A);
    float2*    B    = (float2*)(ws + OFF_B);
    float*     cm   = (float*)(ws + OFF_CM);
    float*     base = (float*)(ws + OFF_BASE);
    _Float16*  h1   = (_Float16*)(ws + OFF_H1);
    _Float16*  h2   = (_Float16*)(ws + OFF_H2);
    float*     scr  = (float*)(ws + OFF_SC);
    float*     acc  = scr;          // [0]
    float*     feat = scr + 64;     // [64..191]
    _Float16*  W2   = (_Float16*)(ws + OFF_W2);
    _Float16*  W3   = (_Float16*)(ws + OFF_W3);
    float*     out  = (float*)d_out;

    // init + weight prep + prog_mean + multipliers
    k_init<<<NFFT / 256, 256, 0, stream>>>(z, A, cm, scr);
    k_wconv<<<(128 * K3TOT + 255) / 256, 256, 0, stream>>>(c2w, c3w, W2, W3);
    k_progmean<<<2048, 256, 0, stream>>>(tw1, tb1, tw2, tb2, tw3, tb3, harm, acc);
    k_bandmult<<<(LTOT + 255) / 256, 256, 0, stream>>>(cm, bw);
    k_specwin<<<1, 32, 0, stream>>>(cm, fw, acc);

    // forward FFT (11 radix-4 Stockham passes; ping-pong stays L2-resident)
    float2 *src = A, *dst = B;
    for (int pp = 0; pp < 11; ++pp) {
        k_fft4<<<NQ / 256, 256, 0, stream>>>(src, dst, 1 << (2 * pp), 2 * pp, 1.0f);
        float2* tmp = src; src = dst; dst = tmp;
    }
    // spectrum in src; apply shaping -> dst (full Hermitian spectrum)
    k_apply<<<(LTOT + 255) / 256, 256, 0, stream>>>(src, dst, cm);
    // inverse FFT
    float2* t2 = src; src = dst; dst = t2;
    for (int pp = 0; pp < 11; ++pp) {
        k_fft4<<<NQ / 256, 256, 0, stream>>>(src, dst, 1 << (2 * pp), 2 * pp, -1.0f);
        float2* tmp = src; src = dst; dst = tmp;
    }
    k_extract<<<(LTOT + 255) / 256, 256, 0, stream>>>(src, base);

    // conv stack
    dim3 g1((LC1 + 255) / 256, 32);
    k_conv1<<<g1, 256, 0, stream>>>(base, c1w, c1b, h1);
    k_conv2<<<(LC2 + 15) / 16, 128, 0, stream>>>(h1, W2, c2b, h2);
    k_conv3<<<(LC3 + 15) / 16, 256, 0, stream>>>(h2, W3, c3b, feat);
    k_final<<<1, 256, 0, stream>>>(feat, mw1, mb1, mw2, mb2, out);

    (void)in_sizes; (void)n_in; (void)out_size; (void)ws_size;
}
